// Decoder_266287972787
// MI455X (gfx1250) — matrix operand order repaired and checked by reading, not compiled, not run on hardware
//
#include <hip/hip_runtime.h>
#include <hip/hip_bf16.h>

#define B 64
#define T_IN 200
#define T_OUT 400
#define NMEL 80
#define PRENET 256
#define RNN 1024
#define GATES 4096
#define ENC 512
#define ATT_DIM 128
#define LOC_F 32
#define LOC_K 31
#define K_ATT (PRENET + ENC + RNN)   /* 1792 */
#define K_DEC (RNN + ENC + RNN)      /* 2560 */
#define TILE_N 128
#define LDS_STRIDE 40                /* 32 bf16 + 8 pad -> 80B row, 16B aligned */

typedef __bf16 bf16_t;
typedef __attribute__((ext_vector_type(16))) __bf16 v16bf;
typedef __attribute__((ext_vector_type(8)))  float  v8f;

union Frag { uint4 u[2]; v16bf v; };

// CDNA5 async global->LDS copy (ASYNCcnt-tracked, no VGPR round trip).
// Builtin signature (from hipcc diagnostic): (v4i AS1*, v4i AS3*, imm int, imm int)
#if defined(__gfx1250__) && __has_builtin(__builtin_amdgcn_global_load_async_to_lds_b128)
#define USE_ASYNC_LDS 1
typedef int v4i_ __attribute__((ext_vector_type(4)));
typedef __attribute__((address_space(1))) v4i_* as1_v4i;
typedef __attribute__((address_space(3))) v4i_* as3_v4i;
#else
#define USE_ASYNC_LDS 0
#endif

__device__ __forceinline__ void async_wait0() {
#if __has_builtin(__builtin_amdgcn_s_wait_asynccnt)
  __builtin_amdgcn_s_wait_asynccnt(0);
#else
  asm volatile("s_wait_asynccnt 0x0" ::: "memory");
#endif
}

__device__ __forceinline__ float sigmoidf_(float x) { return 1.0f / (1.0f + __expf(-x)); }

__device__ __forceinline__ unsigned hash_u32(unsigned x) {
  x ^= x >> 16; x *= 0x7feb352dU; x ^= x >> 15; x *= 0x846ca68bU; x ^= x >> 16;
  return x;
}

// ---------------------------------------------------------------------------
// Stage one K-tile: A 64x32, W 128x32 (bf16) into the given LDS buffers.
// Async path: 3 x global_load_async_to_lds_b128 per thread (lane-addressed).
// Sync fallback: gather all three 128b loads before any ds_store.
// ---------------------------------------------------------------------------
__device__ __forceinline__ void stage_tiles(bf16_t* ldsAbuf, bf16_t* ldsWbuf,
                                            const bf16_t* gA, const bf16_t* gW0,
                                            const bf16_t* gW1,
                                            int row, int chunk, int k0)
{
#if USE_ASYNC_LDS
  __builtin_amdgcn_global_load_async_to_lds_b128(
      (as1_v4i)(gA + k0),  (as3_v4i)(ldsAbuf + row * LDS_STRIDE + chunk), 0, 0);
  __builtin_amdgcn_global_load_async_to_lds_b128(
      (as1_v4i)(gW0 + k0), (as3_v4i)(ldsWbuf + row * LDS_STRIDE + chunk), 0, 0);
  __builtin_amdgcn_global_load_async_to_lds_b128(
      (as1_v4i)(gW1 + k0), (as3_v4i)(ldsWbuf + (row + 64) * LDS_STRIDE + chunk), 0, 0);
#else
  const uint4 ra  = *(const uint4*)(gA + k0);
  const uint4 rw0 = *(const uint4*)(gW0 + k0);
  const uint4 rw1 = *(const uint4*)(gW1 + k0);
  *(uint4*)(ldsAbuf + row * LDS_STRIDE + chunk)        = ra;
  *(uint4*)(ldsWbuf + row * LDS_STRIDE + chunk)        = rw0;
  *(uint4*)(ldsWbuf + (row + 64) * LDS_STRIDE + chunk) = rw1;
  __builtin_prefetch(gW0 + k0 + 32, 0, 1);
  __builtin_prefetch(gW1 + k0 + 32, 0, 1);
#endif
}

// ---------------------------------------------------------------------------
// D[64, N] = A[64, K](bf16) @ W[N, K](bf16)^T,  fp32 accum via WMMA bf16.
// Block: 256 threads = 8 waves; block tile 64(M) x 128(N); K step 32.
// Double-buffered LDS: tile k+1 streams in (async) while tile k is in WMMA.
// ---------------------------------------------------------------------------
__global__ __launch_bounds__(256)
void gemm_bf16_wmma(const bf16_t* __restrict__ A, const bf16_t* __restrict__ W,
                    float* __restrict__ D, int N, int K)
{
  __shared__ __align__(16) bf16_t ldsA[2][64  * LDS_STRIDE];
  __shared__ __align__(16) bf16_t ldsW[2][128 * LDS_STRIDE];

  const int tid    = threadIdx.x;
  const int lane   = tid & 31;
  const int wave   = tid >> 5;
  const int mTile  = wave & 3;
  const int nGroup = wave >> 2;
  const int nBase  = blockIdx.x * TILE_N;

  const int row    = tid >> 2;             // 0..63 staging row
  const int chunk  = (tid & 3) * 8;        // bf16 column offset (16B chunks)

  // Fragment lane mapping per CDNA5 ISA 7.12.2 (wave32):
  // A 16x32 bf16: lanes 0-15 row M=lane, K={0..7,16..23}; lanes 16-31 row M=lane-16, K={8..15,24..31}
  const int a_r  = mTile * 16 + (lane & 15);
  const int a_kb = (lane < 16) ? 0 : 8;
  // B 32x16 bf16: lanes 0-15 col N=lane, K={0..7,16..23}; lanes 16-31 col N=lane-16, K={8..15,24..31}
  const int b_n  = lane & 15;
  const int b_kb = (lane < 16) ? 0 : 8;

  const bf16_t* gA  = &A[(size_t)row * K + chunk];
  const bf16_t* gW0 = &W[(size_t)(nBase + row) * K + chunk];
  const bf16_t* gW1 = &W[(size_t)(nBase + row + 64) * K + chunk];

  v8f zero8 = {0.f, 0.f, 0.f, 0.f, 0.f, 0.f, 0.f, 0.f};
  v8f acc[4] = {zero8, zero8, zero8, zero8};

  // Prologue: stage tile 0
  stage_tiles(ldsA[0], ldsW[0], gA, gW0, gW1, row, chunk, 0);
#if USE_ASYNC_LDS
  async_wait0();
#endif
  __syncthreads();

  int buf = 0;
  for (int k0 = 0; k0 < K; k0 += 32) {
    const int nxt = k0 + 32;
    if (nxt < K)   // overlap next tile's global->LDS traffic with WMMA
      stage_tiles(ldsA[buf ^ 1], ldsW[buf ^ 1], gA, gW0, gW1, row, chunk, nxt);

    const bf16_t* lA = &ldsA[buf][0];
    const bf16_t* lW = &ldsW[buf][0];
    Frag af;
    af.u[0] = *(const uint4*)&lA[a_r * LDS_STRIDE + a_kb];
    af.u[1] = *(const uint4*)&lA[a_r * LDS_STRIDE + a_kb + 16];

#pragma unroll
    for (int nt = 0; nt < 4; ++nt) {
      const bf16_t* wp = &lW[(nGroup * 64 + nt * 16 + b_n) * LDS_STRIDE + b_kb];
      Frag bf;
      bf.u[0] = *(const uint4*)&wp[0];
      bf.u[1] = *(const uint4*)&wp[16];
      acc[nt] = __builtin_amdgcn_wmma_f32_16x16x32_bf16(
          false, af.v, false, bf.v, (short)0, acc[nt], false, false);
    }

    if (nxt < K) {
#if USE_ASYNC_LDS
      async_wait0();   // own wave's async copies done; barrier covers the rest
#endif
      __syncthreads();
    }
    buf ^= 1;
  }

  // C/D 16x16 f32 layout: lanes 0-15 N=lane, M=v; lanes 16-31 N=lane-16, M=8+v
  const int out_m = mTile * 16 + ((lane < 16) ? 0 : 8);
#pragma unroll
  for (int nt = 0; nt < 4; ++nt) {
    const int n = nBase + nGroup * 64 + nt * 16 + (lane & 15);
#pragma unroll
    for (int v = 0; v < 8; ++v)
      D[(size_t)(out_m + v) * N + n] = acc[nt][v];
  }
}

// ---------------------------------------------------------------------------
// Weight conversion fp32 -> bf16, concatenating [W_ih | W_hh] per gate row.
// ---------------------------------------------------------------------------
__global__ void convert_att(const float* __restrict__ wih, const float* __restrict__ whh,
                            bf16_t* __restrict__ Wc)
{
  const size_t total = (size_t)GATES * K_ATT;
  for (size_t idx = (size_t)blockIdx.x * blockDim.x + threadIdx.x; idx < total;
       idx += (size_t)gridDim.x * blockDim.x) {
    const int n = (int)(idx / K_ATT), k = (int)(idx % K_ATT);
    const float w = (k < PRENET + ENC) ? wih[(size_t)n * (PRENET + ENC) + k]
                                       : whh[(size_t)n * RNN + (k - (PRENET + ENC))];
    Wc[idx] = (bf16_t)w;
  }
}

__global__ void convert_dec(const float* __restrict__ wih, const float* __restrict__ whh,
                            bf16_t* __restrict__ Wc)
{
  const size_t total = (size_t)GATES * K_DEC;
  for (size_t idx = (size_t)blockIdx.x * blockDim.x + threadIdx.x; idx < total;
       idx += (size_t)gridDim.x * blockDim.x) {
    const int n = (int)(idx / K_DEC), k = (int)(idx % K_DEC);
    const float w = (k < RNN + ENC) ? wih[(size_t)n * (RNN + ENC) + k]
                                    : whh[(size_t)n * RNN + (k - (RNN + ENC))];
    Wc[idx] = (bf16_t)w;
  }
}

__global__ void zero_bytes(uint4* __restrict__ p, size_t n16)
{
  uint4 z; z.x = 0u; z.y = 0u; z.z = 0u; z.w = 0u;
  for (size_t i = (size_t)blockIdx.x * blockDim.x + threadIdx.x; i < n16;
       i += (size_t)gridDim.x * blockDim.x)
    p[i] = z;
}

// ---------------------------------------------------------------------------
// proc_mem[b,ti,:] = memory[b,ti,:] @ wmem^T   (one block per (b,ti), 128 thr)
// ---------------------------------------------------------------------------
__global__ __launch_bounds__(128)
void procmem_kernel(const float* __restrict__ memory, const float* __restrict__ wmem,
                    float* __restrict__ pmem)
{
  const int blk = blockIdx.x;       // b*T_IN + ti
  const int tid = threadIdx.x;      // d
  __shared__ float s_m[ENC];
  for (int e = tid; e < ENC; e += 128) s_m[e] = memory[(size_t)blk * ENC + e];
  __syncthreads();
  const float* wr = &wmem[tid * ENC];
  float s = 0.f;
  for (int e = 0; e < ENC; ++e) s += s_m[e] * wr[e];
  pmem[(size_t)blk * ATT_DIM + tid] = s;
}

// ---------------------------------------------------------------------------
// Prenet for step t (one block per batch row), writes bf16 into act_att[0:256].
// Dropout uses a deterministic counter hash (no runtime RNG state).
// ---------------------------------------------------------------------------
__global__ __launch_bounds__(256)
void prenet_stage(const float* __restrict__ dec_inputs, const float* __restrict__ w1,
                  const float* __restrict__ w2, bf16_t* __restrict__ act_att, int t)
{
  const int b = blockIdx.x, tid = threadIdx.x;
  __shared__ float s_in[NMEL];
  __shared__ float s_x1[PRENET];
  if (tid < NMEL)
    s_in[tid] = (t == 0) ? 0.f : dec_inputs[((size_t)b * NMEL + tid) * T_OUT + (t - 1)];
  __syncthreads();

  const unsigned seq = (unsigned)(t * B + b) * 256u + (unsigned)tid;
  {
    const float* wr = &w1[tid * NMEL];
    float s = 0.f;
    for (int k = 0; k < NMEL; ++k) s += s_in[k] * wr[k];
    s = fmaxf(s, 0.f);
    s_x1[tid] = (hash_u32(0x9E3779B9u ^ seq) & 1u) ? s * 2.f : 0.f;
  }
  __syncthreads();
  {
    const float* wr = &w2[tid * PRENET];
    float s = 0.f;
    for (int k = 0; k < PRENET; ++k) s += s_x1[k] * wr[k];
    s = fmaxf(s, 0.f);
    s = (hash_u32(0x85EBCA6Bu ^ seq) & 1u) ? s * 2.f : 0.f;
    act_att[(size_t)b * K_ATT + tid] = (bf16_t)s;
  }
}

// ---------------------------------------------------------------------------
// Attention-LSTM pointwise: gates -> (att_h, att_c); stage bf16 activations.
// ---------------------------------------------------------------------------
__global__ __launch_bounds__(256)
void att_cell(const float* __restrict__ gates,
              const float* __restrict__ bih, const float* __restrict__ bhh,
              float* __restrict__ att_h, float* __restrict__ att_c,
              bf16_t* __restrict__ act_dec, bf16_t* __restrict__ act_att)
{
  const int idx = blockIdx.x * blockDim.x + threadIdx.x;   // B*RNN
  const int m = idx >> 10, j = idx & (RNN - 1);
  const float* g = &gates[(size_t)m * GATES];
  const float ig = g[j]            + bih[j]            + bhh[j];
  const float fg = g[RNN + j]      + bih[RNN + j]      + bhh[RNN + j];
  const float gg = g[2 * RNN + j]  + bih[2 * RNN + j]  + bhh[2 * RNN + j];
  const float og = g[3 * RNN + j]  + bih[3 * RNN + j]  + bhh[3 * RNN + j];
  const float cn = sigmoidf_(fg) * att_c[idx] + sigmoidf_(ig) * tanhf(gg);
  const float hn = sigmoidf_(og) * tanhf(cn);
  att_c[idx] = cn;
  att_h[idx] = hn;
  const bf16_t hb = (bf16_t)hn;
  act_dec[(size_t)m * K_DEC + j] = hb;                          // dec GEMM this step
  act_att[(size_t)m * K_ATT + PRENET + ENC + j] = hb;           // att GEMM next step
}

// ---------------------------------------------------------------------------
// Location-sensitive attention: conv + energies + softmax + context.
// One block per batch row; LDS reductions over T_in=200.
// ---------------------------------------------------------------------------
__global__ __launch_bounds__(256)
void attention_kernel(const float* __restrict__ att_h,
                      float* __restrict__ aw, float* __restrict__ awc,
                      float* __restrict__ ctx,
                      const float* __restrict__ pmem,      // [B, T_IN, 128]
                      const float* __restrict__ memory,    // [B, T_IN, 512]
                      const int* __restrict__ mem_len,
                      const float* __restrict__ wq,        // [128, 1024]
                      const float* __restrict__ wv,        // [128]
                      const float* __restrict__ wconv,     // [32, 2, 31]
                      const float* __restrict__ wloc,      // [128, 32]
                      bf16_t* __restrict__ act_dec,
                      bf16_t* __restrict__ act_att,
                      float* __restrict__ align_out,       // [B, T_OUT, T_IN]
                      int t)
{
  const int b = blockIdx.x, tid = threadIdx.x;
  __shared__ float s_h[RNN];
  __shared__ float s_pq[ATT_DIM];
  __shared__ float s_aw[T_IN];
  __shared__ float s_awc[T_IN];
  __shared__ float s_awn[T_IN];
  __shared__ float s_red[256];

  for (int k = tid; k < RNN; k += 256) s_h[k] = att_h[(size_t)b * RNN + k];
  if (tid < T_IN) {
    s_aw[tid]  = aw[b * T_IN + tid];
    s_awc[tid] = awc[b * T_IN + tid];
  }
  __syncthreads();

  if (tid < ATT_DIM) {           // pq = att_h @ wq^T
    const float* wr = &wq[tid * RNN];
    float s = 0.f;
    for (int k = 0; k < RNN; ++k) s += s_h[k] * wr[k];
    s_pq[tid] = s;
  }
  __syncthreads();

  const int mlen = mem_len[b];
  float e_val = -1e9f;
  if (tid < T_IN && tid < mlen) {
    float locf[LOC_F];
    for (int f = 0; f < LOC_F; ++f) {          // location conv, pad 15
      const float* w0 = &wconv[(f * 2 + 0) * LOC_K];
      const float* w1 = &wconv[(f * 2 + 1) * LOC_K];
      float s = 0.f;
      for (int k = 0; k < LOC_K; ++k) {
        const int ti2 = tid + k - LOC_K / 2;
        if (ti2 >= 0 && ti2 < T_IN) s += w0[k] * s_aw[ti2] + w1[k] * s_awc[ti2];
      }
      locf[f] = s;
    }
    const float* pm = &pmem[((size_t)b * T_IN + tid) * ATT_DIM];
    float e = 0.f;
    for (int d = 0; d < ATT_DIM; ++d) {
      const float* wl = &wloc[d * LOC_F];
      float pl = 0.f;
      for (int f = 0; f < LOC_F; ++f) pl += wl[f] * locf[f];
      e += wv[d] * tanhf(pm[d] + s_pq[d] + pl);
    }
    e_val = e;
  }

  // softmax over T_IN (max + sum LDS reductions)
  s_red[tid] = e_val;
  __syncthreads();
  for (int s = 128; s > 0; s >>= 1) {
    if (tid < s) s_red[tid] = fmaxf(s_red[tid], s_red[tid + s]);
    __syncthreads();
  }
  const float emax = s_red[0];
  __syncthreads();
  const float ex = (tid < T_IN) ? __expf(e_val - emax) : 0.f;
  s_red[tid] = ex;
  __syncthreads();
  for (int s = 128; s > 0; s >>= 1) {
    if (tid < s) s_red[tid] += s_red[tid + s];
    __syncthreads();
  }
  const float inv = 1.f / s_red[0];
  __syncthreads();

  if (tid < T_IN) {
    const float awn = ex * inv;
    s_awn[tid] = awn;
    aw[b * T_IN + tid]  = awn;
    awc[b * T_IN + tid] = s_awc[tid] + awn;
    align_out[((size_t)b * T_OUT + t) * T_IN + tid] = awn;
  }
  __syncthreads();

  // ctx = aw_new @ memory[b]
  for (int e2 = tid; e2 < ENC; e2 += 256) {
    float s = 0.f;
    for (int ti = 0; ti < T_IN; ++ti)
      s += s_awn[ti] * memory[((size_t)b * T_IN + ti) * ENC + e2];
    ctx[(size_t)b * ENC + e2] = s;
    const bf16_t cb = (bf16_t)s;
    act_dec[(size_t)b * K_DEC + RNN + e2]    = cb;   // dec GEMM this step
    act_att[(size_t)b * K_ATT + PRENET + e2] = cb;   // att GEMM next step
  }
}

// ---------------------------------------------------------------------------
// Decoder-LSTM pointwise + mel/gate projection (one block per batch row).
// ---------------------------------------------------------------------------
__global__ __launch_bounds__(256)
void dec_cell_proj(const float* __restrict__ gates,
                   const float* __restrict__ bih, const float* __restrict__ bhh,
                   float* __restrict__ dec_h, float* __restrict__ dec_c,
                   const float* __restrict__ ctx,
                   bf16_t* __restrict__ act_dec,
                   const float* __restrict__ wproj, const float* __restrict__ bproj,
                   const float* __restrict__ wgate, const float* __restrict__ bgate,
                   float* __restrict__ mel_out, float* __restrict__ gate_out, int t)
{
  const int b = blockIdx.x, tid = threadIdx.x;
  __shared__ float s_h[RNN];
  __shared__ float s_ctx[ENC];
  const float* g = &gates[(size_t)b * GATES];
  for (int j = tid; j < RNN; j += 256) {
    const float ig = g[j]           + bih[j]           + bhh[j];
    const float fg = g[RNN + j]     + bih[RNN + j]     + bhh[RNN + j];
    const float gg = g[2 * RNN + j] + bih[2 * RNN + j] + bhh[2 * RNN + j];
    const float og = g[3 * RNN + j] + bih[3 * RNN + j] + bhh[3 * RNN + j];
    const float cn = sigmoidf_(fg) * dec_c[(size_t)b * RNN + j] + sigmoidf_(ig) * tanhf(gg);
    const float hn = sigmoidf_(og) * tanhf(cn);
    dec_c[(size_t)b * RNN + j] = cn;
    dec_h[(size_t)b * RNN + j] = hn;
    s_h[j] = hn;
    act_dec[(size_t)b * K_DEC + RNN + ENC + j] = (bf16_t)hn;   // next step
  }
  for (int e = tid; e < ENC; e += 256) s_ctx[e] = ctx[(size_t)b * ENC + e];
  __syncthreads();

  if (tid < NMEL) {
    const float* wr = &wproj[tid * (RNN + ENC)];
    float s = bproj[tid];
    for (int k = 0; k < RNN; ++k) s += s_h[k] * wr[k];
    for (int k = 0; k < ENC; ++k) s += s_ctx[k] * wr[RNN + k];
    mel_out[((size_t)b * NMEL + tid) * T_OUT + t] = s;
  } else if (tid == NMEL) {
    float s = bgate[0];
    for (int k = 0; k < RNN; ++k) s += s_h[k] * wgate[k];
    for (int k = 0; k < ENC; ++k) s += s_ctx[k] * wgate[RNN + k];
    gate_out[(size_t)b * T_OUT + t] = s;
  }
}

// ---------------------------------------------------------------------------
extern "C" void kernel_launch(void* const* d_in, const int* in_sizes, int n_in,
                              void* d_out, int out_size, void* d_ws, size_t ws_size,
                              hipStream_t stream)
{
  (void)in_sizes; (void)n_in; (void)out_size; (void)ws_size;
  const float* memory         = (const float*)d_in[0];
  const float* decoder_inputs = (const float*)d_in[1];
  const int*   memory_lengths = (const int*)  d_in[2];
  const float* w_prenet1      = (const float*)d_in[3];
  const float* w_prenet2      = (const float*)d_in[4];
  const float* wih_att        = (const float*)d_in[5];
  const float* whh_att        = (const float*)d_in[6];
  const float* bih_att        = (const float*)d_in[7];
  const float* bhh_att        = (const float*)d_in[8];
  const float* wq             = (const float*)d_in[9];
  const float* wmem           = (const float*)d_in[10];
  const float* wv             = (const float*)d_in[11];
  const float* wconv          = (const float*)d_in[12];
  const float* wloc           = (const float*)d_in[13];
  const float* wih_dec        = (const float*)d_in[14];
  const float* whh_dec        = (const float*)d_in[15];
  const float* bih_dec        = (const float*)d_in[16];
  const float* bhh_dec        = (const float*)d_in[17];
  const float* wproj          = (const float*)d_in[18];
  const float* bproj          = (const float*)d_in[19];
  const float* wgate          = (const float*)d_in[20];
  const float* bgate          = (const float*)d_in[21];

  float* mel_out   = (float*)d_out;                               // [B, 80, T_OUT]
  float* gate_out  = mel_out + (size_t)B * NMEL * T_OUT;          // [B, T_OUT]
  float* align_out = gate_out + (size_t)B * T_OUT;                // [B, T_OUT, T_IN]

  char* wsb = (char*)d_ws;
  size_t off = 0;
  auto carve = [&](size_t bytes) -> void* {
    void* p = wsb + off;
    off += (bytes + 255) & ~(size_t)255;
    return p;
  };

  bf16_t* Watt = (bf16_t*)carve((size_t)GATES * K_ATT * sizeof(bf16_t));  // 14.0 MB
  bf16_t* Wdec = (bf16_t*)carve((size_t)GATES * K_DEC * sizeof(bf16_t));  // 20.0 MB
  float*  pmem = (float*) carve((size_t)B * T_IN * ATT_DIM * sizeof(float));

  char* state0 = wsb + off;                                       // zeroed region
  float*  att_h   = (float*) carve((size_t)B * RNN * sizeof(float));
  float*  att_c   = (float*) carve((size_t)B * RNN * sizeof(float));
  float*  dec_h   = (float*) carve((size_t)B * RNN * sizeof(float));
  float*  dec_c   = (float*) carve((size_t)B * RNN * sizeof(float));
  float*  aw      = (float*) carve((size_t)B * T_IN * sizeof(float));
  float*  awc     = (float*) carve((size_t)B * T_IN * sizeof(float));
  float*  ctx     = (float*) carve((size_t)B * ENC * sizeof(float));
  bf16_t* act_att = (bf16_t*)carve((size_t)B * K_ATT * sizeof(bf16_t));
  bf16_t* act_dec = (bf16_t*)carve((size_t)B * K_DEC * sizeof(bf16_t));
  const size_t state_bytes = (size_t)((wsb + off) - state0);

  float* gates_att = (float*)carve((size_t)B * GATES * sizeof(float));
  float* gates_dec = (float*)carve((size_t)B * GATES * sizeof(float));

  // ---- one-time setup (weights fit in L2; converted to bf16 once per launch)
  convert_att<<<2048, 256, 0, stream>>>(wih_att, whh_att, Watt);
  convert_dec<<<2048, 256, 0, stream>>>(wih_dec, whh_dec, Wdec);
  zero_bytes<<<512, 256, 0, stream>>>((uint4*)state0, state_bytes / 16);
  procmem_kernel<<<B * T_IN, 128, 0, stream>>>(memory, wmem, pmem);

  // ---- sequential decoder scan
  for (int t = 0; t < T_OUT; ++t) {
    prenet_stage<<<B, 256, 0, stream>>>(decoder_inputs, w_prenet1, w_prenet2, act_att, t);
    gemm_bf16_wmma<<<GATES / TILE_N, 256, 0, stream>>>(act_att, Watt, gates_att, GATES, K_ATT);
    att_cell<<<(B * RNN) / 256, 256, 0, stream>>>(gates_att, bih_att, bhh_att,
                                                  att_h, att_c, act_dec, act_att);
    attention_kernel<<<B, 256, 0, stream>>>(att_h, aw, awc, ctx, pmem, memory, memory_lengths,
                                            wq, wv, wconv, wloc, act_dec, act_att, align_out, t);
    gemm_bf16_wmma<<<GATES / TILE_N, 256, 0, stream>>>(act_dec, Wdec, gates_dec, GATES, K_DEC);
    dec_cell_proj<<<B, 256, 0, stream>>>(gates_dec, bih_dec, bhh_dec, dec_h, dec_c, ctx, act_dec,
                                         wproj, bproj, wgate, bgate, mel_out, gate_out, t);
  }
}
